// WireframeDetector_292057776220
// MI455X (gfx1250) — matrix operand order repaired and checked by reading, not compiled, not run on hardware
//
#include <hip/hip_runtime.h>

// ---------------------------------------------------------------------------
// Types for CDNA5 WMMA
// ---------------------------------------------------------------------------
typedef _Float16 v16h __attribute__((ext_vector_type(16)));
typedef _Float16 v8h  __attribute__((ext_vector_type(8)));
typedef float    v8f  __attribute__((ext_vector_type(8)));

#define HW_  16384   // 128*128
#define WID_ 128
#define NJ_  300
#define NL_  5000
#define NLP_ 5008    // padded to multiple of 16
#define KD_  1024

__device__ __forceinline__ float sigmoidf_(float x) { return 1.0f / (1.0f + expf(-x)); }

// ---------------------------------------------------------------------------
// 1) Head decode: jloc = sigmoid(c2-c1) with 3x3 NMS, cloc = sigmoid(c6-c5)
// ---------------------------------------------------------------------------
__global__ void decode_kernel(const float* __restrict__ out,
                              float* __restrict__ nmsJ, float* __restrict__ clocB) {
    int idx = blockIdx.x * blockDim.x + threadIdx.x;
    if (idx >= HW_) return;
    int y = idx >> 7, x = idx & 127;
    const float* c1 = out + 1 * HW_;
    const float* c2 = out + 2 * HW_;
    float jl = 0.0f, mx = -1e30f;
    for (int dy = -1; dy <= 1; ++dy) {
        int yy = y + dy; if (yy < 0 || yy > 127) continue;
        for (int dx = -1; dx <= 1; ++dx) {
            int xx = x + dx; if (xx < 0 || xx > 127) continue;
            int q = yy * WID_ + xx;
            float v = sigmoidf_(c2[q] - c1[q]);
            if (q == idx) jl = v;
            mx = fmaxf(mx, v);
        }
    }
    nmsJ[idx]  = (jl == mx) ? jl : 0.0f;
    clocB[idx] = sigmoidf_(out[6 * HW_ + idx] - out[5 * HW_ + idx]);
}

// ---------------------------------------------------------------------------
// 2) Full bitonic sort of 16384 scores in LDS (single workgroup, 128 KB LDS).
//    Composite key: descending score (non-negative -> monotone bits), then
//    ascending index (lax.top_k tie order). Emits first K indices.
// ---------------------------------------------------------------------------
__global__ __launch_bounds__(1024)
void topk_sort_kernel(const float* __restrict__ score, int* __restrict__ out_idx, int K) {
    __shared__ unsigned long long s[HW_];   // 128 KB (CDNA5: up to 320 KB / WG)
    const int tid = threadIdx.x;
    for (int i = tid; i < HW_; i += 1024) {
        unsigned int bits = __float_as_uint(score[i]);       // scores are >= 0
        s[i] = ((unsigned long long)(bits ^ 0xFFFFFFFFu) << 32) | (unsigned int)i;
    }
    __syncthreads();
    for (int k = 2; k <= HW_; k <<= 1) {
        for (int j = k >> 1; j > 0; j >>= 1) {
            for (int i = tid; i < HW_; i += 1024) {
                int ixj = i ^ j;
                if (ixj > i) {
                    unsigned long long a = s[i], b = s[ixj];
                    bool asc = ((i & k) == 0);
                    if ((a > b) == asc) { s[i] = b; s[ixj] = a; }
                }
            }
            __syncthreads();
        }
    }
    for (int i = tid; i < K; i += 1024) out_idx[i] = (int)(s[i] & 0xFFFFFFFFu);
}

// ---------------------------------------------------------------------------
// 3) Junction coordinates from top-300 indices
// ---------------------------------------------------------------------------
__global__ void junc_kernel(const float* __restrict__ out, const int* __restrict__ topJ,
                            float* __restrict__ junc) {
    int t = blockIdx.x * blockDim.x + threadIdx.x;
    if (t >= NJ_) return;
    int idx = topJ[t];
    junc[t * 2 + 0] = (float)(idx & 127) + out[3 * HW_ + idx] + 0.5f;
    junc[t * 2 + 1] = (float)(idx >> 7)  + out[4 * HW_ + idx] + 0.5f;
}

// ---------------------------------------------------------------------------
// 4) Line endpoints from top-5000 center indices
//    endpoint p: x = cx + tanh(ch[9+p])*128, y = cy + tanh(ch[11+p])*128
// ---------------------------------------------------------------------------
__global__ void line_kernel(const float* __restrict__ out, const int* __restrict__ topL,
                            float* __restrict__ lineEP) {
    int t = blockIdx.x * blockDim.x + threadIdx.x;
    if (t >= NL_) return;
    int idx = topL[t];
    float cx = (float)(idx & 127) + out[7 * HW_ + idx] + 0.5f;
    float cy = (float)(idx >> 7)  + out[8 * HW_ + idx] + 0.5f;
    float4 v;
    v.x = cx + tanhf(out[ 9 * HW_ + idx]) * 128.0f;   // e0.x
    v.y = cy + tanhf(out[11 * HW_ + idx]) * 128.0f;   // e0.y
    v.z = cx + tanhf(out[10 * HW_ + idx]) * 128.0f;   // e1.x
    v.w = cy + tanhf(out[12 * HW_ + idx]) * 128.0f;   // e1.y
    *(float4*)(lineEP + t * 4) = v;
}

// ---------------------------------------------------------------------------
// 5) Match endpoints to nearest junctions, order by y, write LOI + iskeep
// ---------------------------------------------------------------------------
__global__ __launch_bounds__(256)
void match_kernel(const float* __restrict__ junc, const float* __restrict__ lineEP,
                  float* __restrict__ loi_out, float* __restrict__ iskeep) {
    __shared__ float ju[NJ_ * 2];
    for (int i = threadIdx.x; i < NJ_ * 2; i += 256) ju[i] = junc[i];
    __syncthreads();
    int t = blockIdx.x * 256 + threadIdx.x;
    if (t >= NL_) return;
    float4 e = *(const float4*)(lineEP + t * 4);
    float b1 = 1e30f, b2 = 1e30f; int i1 = 0, i2 = 0;
    for (int j = 0; j < NJ_; ++j) {
        float jx = ju[2 * j], jy = ju[2 * j + 1];
        float dx = e.x - jx, dy = e.y - jy;
        float d = dx * dx + dy * dy;
        if (d < b1) { b1 = d; i1 = j; }
        dx = e.z - jx; dy = e.w - jy;
        d = dx * dx + dy * dy;
        if (d < b2) { b2 = d; i2 = j; }
    }
    int imin = min(i1, i2), imax = max(i1, i2);
    iskeep[t] = (imin != imax) ? 1.0f : 0.0f;
    float p0x = ju[2 * imin], p0y = ju[2 * imin + 1];
    float p1x = ju[2 * imax], p1y = ju[2 * imax + 1];
    if (p0y > p1y) { float tx = p0x; p0x = p1x; p1x = tx; float ty = p0y; p0y = p1y; p1y = ty; }
    float4 v = {p0x, p0y, p1x, p1y};
    *(float4*)(loi_out + t * 4) = v;
}

// ---------------------------------------------------------------------------
// 6) LOI pooling: 32 bilinear samples per line, MaxPool1d(4) -> f16 features
//    Feature index = c*8 + point_group. One block per line, thread = channel.
// ---------------------------------------------------------------------------
__global__ __launch_bounds__(128)
void loipool_kernel(const float* __restrict__ feat, const float* __restrict__ loi_out,
                    _Float16* __restrict__ F) {
    int n = blockIdx.x;
    int c = threadIdx.x;
    float4 L = *(const float4*)(loi_out + n * 4);   // p0x,p0y,p1x,p1y
    const float* fc_ = feat + (long)c * HW_;
    float mv[8];
#pragma unroll
    for (int q = 0; q < 8; ++q) mv[q] = -1e30f;
#pragma unroll
    for (int p = 0; p < 32; ++p) {
        float tt = (float)p * (1.0f / 31.0f);
        float px = (1.0f - tt) * L.x + tt * L.z - 0.5f;
        float py = (1.0f - tt) * L.y + tt * L.w - 0.5f;
        float px0 = fminf(fmaxf(floorf(px), 0.0f), 127.0f);
        float py0 = fminf(fmaxf(floorf(py), 0.0f), 127.0f);
        float px1 = fminf(px0 + 1.0f, 127.0f);
        float py1 = fminf(py0 + 1.0f, 127.0f);
        int x0 = (int)px0, y0 = (int)py0, x1 = (int)px1, y1 = (int)py1;
        float v = fc_[y0 * WID_ + x0] * (py1 - py) * (px1 - px)
                + fc_[y1 * WID_ + x0] * (py - py0) * (px1 - px)
                + fc_[y0 * WID_ + x1] * (py1 - py) * (px - px0)
                + fc_[y1 * WID_ + x1] * (py - py0) * (px - px0);
        mv[p >> 2] = fmaxf(mv[p >> 2], v);
    }
    v8h pk;
#pragma unroll
    for (int q = 0; q < 8; ++q) pk[q] = (_Float16)mv[q];
    *(v8h*)(F + (long)n * KD_ + c * 8) = pk;   // 16B aligned contiguous store
}

// ---------------------------------------------------------------------------
// 7) Weight convert + transpose: f32 [K,N] row-major -> f16 [N,K] row-major
// ---------------------------------------------------------------------------
__global__ void wconv_kernel(const float* __restrict__ W, _Float16* __restrict__ Wt,
                             int K, int N) {
    int id = blockIdx.x * blockDim.x + threadIdx.x;
    if (id >= K * N) return;
    int k = id / N, n = id % N;
    Wt[(long)n * K + k] = (_Float16)W[id];
}

// ---------------------------------------------------------------------------
// 8) WMMA GEMM: H = relu((A[Mp,K] @ B[K,N] + bias) * scale + shift), f16 out.
//    A f16 row-major, Bt f16 = B transposed [N,K] row-major.
//    Block = 8 waves; wave w owns N-strip [w*128, w*128+128) = 8 x 16x16 tiles;
//    grid.x = M/16 tiles. 256 v_wmma_f32_16x16x32_f16 per wave.
// ---------------------------------------------------------------------------
__global__ __launch_bounds__(256)
void gemm_wmma_kernel(const _Float16* __restrict__ A, const _Float16* __restrict__ Bt,
                      const float* __restrict__ bias, const float* __restrict__ scale,
                      const float* __restrict__ shift, _Float16* __restrict__ Hout,
                      int K, int N) {
    const int mtile = blockIdx.x;
    const int lane  = threadIdx.x & 31;
    const int wave  = threadIdx.x >> 5;
    const int lo = lane & 15;
    const int hi = lane >> 4;
    const long arow = (long)(mtile * 16 + lo) * K;
    const int nbase = wave * 128;

    v8f acc[8];
#pragma unroll
    for (int t = 0; t < 8; ++t)
#pragma unroll
        for (int e = 0; e < 8; ++e) acc[t][e] = 0.0f;

    for (int kk = 0; kk < K; kk += 32) {
        // A tile (16x32 f16): lane holds row (lane&15),
        // K chunks [kk + hi*16, +8) and [kk + hi*16 + 8, +8) -> two b128 loads.
        const int b0 = kk + (hi ? 16 : 0);
        const int b1 = kk + (hi ? 24 : 8);
        v8h alo = *(const v8h*)(A + arow + b0);
        v8h ahi = *(const v8h*)(A + arow + b1);
        v16h a;
#pragma unroll
        for (int i = 0; i < 8; ++i) { a[i] = alo[i]; a[i + 8] = ahi[i]; }
#pragma unroll
        for (int nt = 0; nt < 8; ++nt) {
            // B tile (32x16): lane holds column (lane&15), K = kk + hi*16 .. +16
            const int n = nbase + nt * 16 + lo;
            v16h b = *(const v16h*)(Bt + (long)n * K + kk + (hi ? 16 : 0));
            acc[nt] = __builtin_amdgcn_wmma_f32_16x16x32_f16(
                false, a, false, b, (short)0, acc[nt], false, false);
        }
    }
    // Epilogue: D layout -> VGPR r: row = mtile*16 + r + hi*8, col = nbase+nt*16+lo
#pragma unroll
    for (int nt = 0; nt < 8; ++nt) {
        const int n = nbase + nt * 16 + lo;
        const float bs = bias[n], sc = scale[n], sh = shift[n];
#pragma unroll
        for (int r = 0; r < 8; ++r) {
            int mm = mtile * 16 + r + (hi ? 8 : 0);
            float v = (acc[nt][r] + bs) * sc + sh;
            v = v > 0.0f ? v : 0.0f;
            Hout[(long)mm * N + n] = (_Float16)v;
        }
    }
}

// ---------------------------------------------------------------------------
// 9) fc3 (1024 -> 1) + sigmoid * iskeep. One wave per line.
// ---------------------------------------------------------------------------
__global__ __launch_bounds__(256)
void fc3_kernel(const _Float16* __restrict__ H2, const float* __restrict__ w3,
                const float* __restrict__ b3, const float* __restrict__ iskeep,
                float* __restrict__ score_out) {
    int wave = threadIdx.x >> 5, lane = threadIdx.x & 31;
    int m = blockIdx.x * 8 + wave;
    if (m >= NL_) return;                      // wave-uniform exit
    const _Float16* row = H2 + (long)m * KD_;
    float s = 0.0f;
#pragma unroll 4
    for (int i = 0; i < 32; ++i) {
        int n = i * 32 + lane;
        s += (float)row[n] * w3[n];
    }
    for (int off = 16; off; off >>= 1) s += __shfl_xor(s, off, 32);
    if (lane == 0) {
        float sc = 1.0f / (1.0f + expf(-(s + b3[0])));
        score_out[m] = sc * iskeep[m];
    }
}

// ---------------------------------------------------------------------------
// Launcher
// ---------------------------------------------------------------------------
extern "C" void kernel_launch(void* const* d_in, const int* in_sizes, int n_in,
                              void* d_out, int out_size, void* d_ws, size_t ws_size,
                              hipStream_t stream) {
    const float* outputs = (const float*)d_in[0];
    const float* feats   = (const float*)d_in[1];
    const float* fc1_w   = (const float*)d_in[2];
    const float* fc1_b   = (const float*)d_in[3];
    const float* bn1s    = (const float*)d_in[4];
    const float* bn1h    = (const float*)d_in[5];
    const float* fc2_w   = (const float*)d_in[6];
    const float* fc2_b   = (const float*)d_in[7];
    const float* bn2s    = (const float*)d_in[8];
    const float* bn2h    = (const float*)d_in[9];
    const float* fc3_w   = (const float*)d_in[10];
    const float* fc3_b   = (const float*)d_in[11];
    float* out = (float*)d_out;

    const int B = in_sizes[0] / (13 * HW_);

    // Carve workspace (256B aligned).
    uintptr_t p = (uintptr_t)d_ws;
    auto carve = [&](size_t bytes) -> void* {
        p = (p + 255) & ~(uintptr_t)255;
        void* r = (void*)p; p += bytes; return r;
    };
    _Float16* W1t   = (_Float16*)carve((size_t)KD_ * KD_ * 2);
    _Float16* W2t   = (_Float16*)carve((size_t)KD_ * KD_ * 2);
    float*    nmsJ  = (float*)carve(HW_ * 4);
    float*    clocB = (float*)carve(HW_ * 4);
    int*      topJ  = (int*)carve(NJ_ * 4);
    int*      topL  = (int*)carve(NL_ * 4);
    float*    junc  = (float*)carve(NJ_ * 2 * 4);
    float*    lineEP= (float*)carve(NL_ * 4 * 4);
    float*    iskeep= (float*)carve(NL_ * 4);
    _Float16* F     = (_Float16*)carve((size_t)NLP_ * KD_ * 2);
    _Float16* H1    = (_Float16*)carve((size_t)NLP_ * KD_ * 2);
    _Float16* H2    = (_Float16*)carve((size_t)NLP_ * KD_ * 2);

    // Weights: convert+transpose once per launch.
    wconv_kernel<<<(KD_ * KD_ + 255) / 256, 256, 0, stream>>>(fc1_w, W1t, KD_, KD_);
    wconv_kernel<<<(KD_ * KD_ + 255) / 256, 256, 0, stream>>>(fc2_w, W2t, KD_, KD_);

    for (int b = 0; b < B; ++b) {
        const float* ob = outputs + (size_t)b * 13 * HW_;
        const float* fb = feats   + (size_t)b * 128 * HW_;
        float* loi_b   = out + (size_t)b * NL_ * 4;
        float* score_b = out + (size_t)B * NL_ * 4 + (size_t)b * NL_;

        decode_kernel<<<HW_ / 256, 256, 0, stream>>>(ob, nmsJ, clocB);
        topk_sort_kernel<<<1, 1024, 0, stream>>>(nmsJ, topJ, NJ_);
        topk_sort_kernel<<<1, 1024, 0, stream>>>(clocB, topL, NL_);
        junc_kernel<<<2, 256, 0, stream>>>(ob, topJ, junc);
        line_kernel<<<(NL_ + 255) / 256, 256, 0, stream>>>(ob, topL, lineEP);
        match_kernel<<<(NL_ + 255) / 256, 256, 0, stream>>>(junc, lineEP, loi_b, iskeep);
        loipool_kernel<<<NL_, 128, 0, stream>>>(fb, loi_b, F);
        gemm_wmma_kernel<<<NLP_ / 16, 256, 0, stream>>>(F,  W1t, fc1_b, bn1s, bn1h, H1, KD_, KD_);
        gemm_wmma_kernel<<<NLP_ / 16, 256, 0, stream>>>(H1, W2t, fc2_b, bn2s, bn2h, H2, KD_, KD_);
        fc3_kernel<<<(NL_ + 7) / 8, 256, 0, stream>>>(H2, fc3_w, fc3_b, iskeep, score_b);
    }
}
